// TwoStreamRelativeAttention_53034256171182
// MI455X (gfx1250) — matrix operand order repaired and checked by reading, not compiled, not run on hardware
//
#include <hip/hip_runtime.h>
#include <hip/hip_bf16.h>
#include <cstdint>

// Problem constants (from reference)
#define SLEN  1024
#define BDIM  2
#define DDIM  1024
#define NHEAD 16
#define HDIM  64
#define RLEN  2048
#define SCALE_F 0.125f   // 1/sqrt(64)

typedef __bf16 bf16;
typedef __attribute__((ext_vector_type(16))) __bf16 v16bf;
typedef __attribute__((ext_vector_type(8)))  float  v8f;

struct U4x2 { uint4 lo, hi; };

// ---- WMMA helper (CDNA5 v_wmma_f32_16x16x32_bf16) ---------------------------
__device__ __forceinline__ v8f wmma_bf16(v16bf a, v16bf b, v8f c) {
  return __builtin_amdgcn_wmma_f32_16x16x32_bf16(false, a, false, b, (short)0, c,
                                                 false, false);
}
// A fragment = two contiguous 16B runs (callers pass p, p+8: k = e+(L>=16?16:0))
__device__ __forceinline__ v16bf frag_from2(const bf16* p0, const bf16* p1) {
  U4x2 t{*(const uint4*)p0, *(const uint4*)p1};
  return __builtin_bit_cast(v16bf, t);
}
// B fragment stored contiguously (fragment-major LDS): one aligned 32B load
__device__ __forceinline__ v16bf frag32(const bf16* p) {
  return *(const v16bf*)p;
}
__device__ __forceinline__ void store4bf(bf16* dst, float4 v) {
  union { bf16 b[4]; uint2 u; } t;
  t.b[0] = (bf16)v.x; t.b[1] = (bf16)v.y; t.b[2] = (bf16)v.z; t.b[3] = (bf16)v.w;
  *(uint2*)dst = t.u;
}
// ---- CDNA5 async global->LDS (16B per lane), ASYNCcnt-tracked ---------------
__device__ __forceinline__ void async_b128(unsigned lds_byte, const void* g) {
  asm volatile("global_load_async_to_lds_b128 %0, %1, off"
               :: "v"(lds_byte), "v"(g) : "memory");
}
__device__ __forceinline__ void wait_async0() {
  asm volatile("s_wait_asynccnt 0x0" ::: "memory");
}
__device__ __forceinline__ unsigned lds_off(const void* p) {
  return (unsigned)(unsigned long long)p;  // low 32 bits = LDS offset
}
// B-fragment-major permutation: element (row j, col d0..d0+7, 32-wide k-steps)
// -> slot (((j>>4)*2 + d0>>5)*32 + (j&15) + 16*((d0&31)>=16))*16 + (d0&15)
__device__ __forceinline__ int frag_off(int j, int d0) {  // d0 multiple of 8
  int t = j >> 4, n = j & 15;
  int ks = d0 >> 5, rem = d0 & 31;
  int Lhi = rem >> 4, e0 = rem & 15;
  return (((t * 2 + ks) * 32) + n + 16 * Lhi) * 16 + e0;
}

// ============================================================================
// fp32 [D][NH] -> bf16 [NH][D]  (transpose+convert, for q/k/v/r weights)
// ============================================================================
__global__ void transpose_w_kernel(const float* __restrict__ W,
                                   bf16* __restrict__ WT) {
  int idx = blockIdx.x * blockDim.x + threadIdx.x;
  if (idx >= DDIM * NHEAD * HDIM) return;
  int r = idx >> 10, c = idx & 1023;
  WT[(size_t)c * DDIM + r] = (bf16)W[idx];
}
// fp32 -> bf16 same layout (proj_o is already [hD][k])
__global__ void convert_w_kernel(const float* __restrict__ W,
                                 bf16* __restrict__ WC) {
  int idx = blockIdx.x * blockDim.x + threadIdx.x;
  if (idx >= DDIM * NHEAD * HDIM) return;
  WC[idx] = (bf16)W[idx];
}

// ============================================================================
// Projection GEMM: OUT[b][n][row][h] = sum_d X[row][d] * W[d][nh] (+bias)
// X fp32 [Srows*B][D]; WT bf16 [nh][D] (pre-transposed).  128 thr / 4 waves,
// 64x64 tile, K-step 32.  outT writes a per-head transposed copy (for V).
// ============================================================================
__global__ __launch_bounds__(128)
void proj_gemm_kernel(const float* __restrict__ X, const bf16* __restrict__ WT,
                      int Srows,
                      bf16* __restrict__ out0, const float* __restrict__ bias0,
                      bf16* __restrict__ out1, const float* __restrict__ bias1,
                      bf16* __restrict__ out2, const float* __restrict__ bias2,
                      bf16* __restrict__ outT) {
  __shared__ __align__(32) bf16 ldsX[64 * 32];  // [m][k] row-major
  __shared__ __align__(32) bf16 ldsW[64 * 32];  // [n_local][k] row-major
  const int tid  = threadIdx.x;
  const int wave = tid >> 5, lane = tid & 31;
  const int m0 = blockIdx.x * 64, n0 = blockIdx.y * 64;
  const int Lhi8  = (lane >= 16) ? 8 : 0;
  const int Lhi16 = (lane >= 16) ? 16 : 0;
  const int ml = 16 * wave + (lane & 15);

  v8f acc[4];
#pragma unroll
  for (int t = 0; t < 4; ++t)
#pragma unroll
    for (int z = 0; z < 8; ++z) acc[t][z] = 0.f;

  for (int k0 = 0; k0 < DDIM; k0 += 32) {
    for (int idx = tid; idx < 512; idx += 128) {   // X tile 64x32 fp32->bf16
      int r2 = idx >> 3, c4 = (idx & 7) * 4;
      float4 vx = *(const float4*)&X[(size_t)(m0 + r2) * DDIM + k0 + c4];
      store4bf(&ldsX[r2 * 32 + c4], vx);
    }
    for (int idx = tid; idx < 256; idx += 128) {   // W tile 64 cols x 32 k
      int nl = idx >> 2, c8 = (idx & 3) * 8;
      *(uint4*)&ldsW[nl * 32 + c8] =
          *(const uint4*)&WT[(size_t)(n0 + nl) * DDIM + k0 + c8];
    }
    __syncthreads();
    v16bf a;
    {
      const bf16* p = &ldsX[ml * 32 + Lhi16];
      a = frag_from2(p, p + 8);
    }
#pragma unroll
    for (int t = 0; t < 4; ++t)
      acc[t] = wmma_bf16(a, frag32(&ldsW[(16 * t + (lane & 15)) * 32 + Lhi16]),
                         acc[t]);
    __syncthreads();
  }
  // epilogue: scatter to per-head [b][n][row][h] (+optional transposed copy)
#pragma unroll
  for (int t = 0; t < 4; ++t) {
#pragma unroll
    for (int v = 0; v < 8; ++v) {
      int m  = m0 + 16 * wave + v + ((lane >= 16) ? 8 : 0);
      int nh = n0 + 16 * t + (lane & 15);
      int i = m / BDIM, bb = m % BDIM;
      int n = nh >> 6, hh = nh & 63;
      size_t o = (((size_t)(bb * NHEAD + n)) * Srows + i) * HDIM + hh;
      float val = acc[t][v];
      if (out0) out0[o] = (bf16)(val + (bias0 ? bias0[nh] : 0.f));
      if (out1) out1[o] = (bf16)(val + (bias1 ? bias1[nh] : 0.f));
      if (out2) out2[o] = (bf16)(val + (bias2 ? bias2[nh] : 0.f));
      if (outT)
        outT[((size_t)(bb * NHEAD + n) * HDIM + hh) * Srows + i] = (bf16)val;
    }
  }
}

// ============================================================================
// ef[b][n][i][s] = sum_h qsb[b][n][i][h] * seg_embed[s][n][h]
// ============================================================================
__global__ void ef_kernel(const bf16* __restrict__ qsb,
                          const float* __restrict__ seg_embed,
                          float* __restrict__ ef) {
  int idx = blockIdx.x * blockDim.x + threadIdx.x;  // over B*N*S
  if (idx >= BDIM * NHEAD * SLEN) return;
  int bn = idx / SLEN;
  int n  = bn % NHEAD;
  const bf16* q = qsb + (size_t)idx * HDIM;
  float a0 = 0.f, a1 = 0.f;
  for (int hh = 0; hh < HDIM; ++hh) {
    float qv = (float)q[hh];
    a0 += qv * seg_embed[(0 * NHEAD + n) * HDIM + hh];
    a1 += qv * seg_embed[(1 * NHEAD + n) * HDIM + hh];
  }
  ef[(size_t)idx * 2 + 0] = a0;
  ef[(size_t)idx * 2 + 1] = a1;
}

// ============================================================================
// Flash-style relative attention core.  128 threads / 4 waves, 64-row i-tile,
// causal loop over 64-key j-tiles.  K/Kr staged fragment-major and V staged
// pre-transposed via global_load_async_to_lds_b128; rel_shift handled by the
// 64x128 BD tile (row li reads col (63-li)+jl).
// ============================================================================
__global__ __launch_bounds__(128)
void attn_kernel(const bf16* __restrict__ q_ac, const bf16* __restrict__ q_bd,
                 const bf16* __restrict__ Kh,   const bf16* __restrict__ vT,
                 const bf16* __restrict__ Kr,   const float* __restrict__ ef,
                 const unsigned char* __restrict__ seg_mat,
                 const float* __restrict__ mask,
                 bf16* __restrict__ attn_out) {
  __shared__ __align__(32) bf16 sKP[64 * 64];   // K frag-major, then P row-major
  __shared__ __align__(32) bf16 sVt[64 * 64];   // [d][j] row-major
  __shared__ __align__(32) bf16 sKr[128 * 64];  // kr window, frag-major
  __shared__ __align__(32) bf16 sBD[64 * 128];  // BD scores row-major
  __shared__ __align__(32) bf16 sS[64 * 64];    // AC scores row-major
  __shared__ float sM[64], sL[64], sFac[64];

  const int tid = threadIdx.x, wave = tid >> 5, lane = tid & 31;
  const int i0 = blockIdx.x * 64;
  const int bn = blockIdx.y;
  const int b  = bn / NHEAD;
  const size_t headoff = (size_t)bn * SLEN * HDIM;
  const bf16* Kp  = Kh + headoff;
  const bf16* Vp  = vT + (size_t)bn * HDIM * SLEN;
  const bf16* Krp = Kr + (size_t)bn * RLEN * HDIM;
  const int Lhi8  = (lane >= 16) ? 8 : 0;
  const int Lhi16 = (lane >= 16) ? 16 : 0;
  const unsigned offK = lds_off(sKP), offV = lds_off(sVt), offKr = lds_off(sKr);

  if (tid < 64) { sM[tid] = -3.0e38f; sL[tid] = 0.f; }

  // Q fragments (A operands) for this wave's 16 rows, both bias variants
  v16bf a_ac[2], a_bd[2];
  {
    const int mrow = i0 + 16 * wave + (lane & 15);
    const bf16* qa = q_ac + headoff + (size_t)mrow * HDIM;
    const bf16* qb = q_bd + headoff + (size_t)mrow * HDIM;
#pragma unroll
    for (int ks = 0; ks < 2; ++ks) {
      a_ac[ks] = frag_from2(qa + Lhi16 + 32 * ks, qa + Lhi16 + 32 * ks + 8);
      a_bd[ks] = frag_from2(qb + Lhi16 + 32 * ks, qb + Lhi16 + 32 * ks + 8);
    }
  }
  v8f o[4];
#pragma unroll
  for (int t = 0; t < 4; ++t)
#pragma unroll
    for (int z = 0; z < 8; ++z) o[t][z] = 0.f;
  __syncthreads();

  for (int j0 = 0; j0 <= i0; j0 += 64) {
    // ---- async staging: K (frag-major), Vt (row-major), Kr window (frag-major)
    {
      const bf16* gK = Kp + (size_t)j0 * HDIM;
      for (int idx = tid; idx < 512; idx += 128) {
        int j = idx >> 3, d0 = (idx & 7) * 8;
        async_b128(offK + (unsigned)frag_off(j, d0) * 2, gK + idx * 8);
      }
      const bf16* gV = Vp + j0;  // rows d, stride SLEN
      for (int idx = tid; idx < 512; idx += 128) {
        int r2 = idx >> 3, c0 = (idx & 7) * 8;
        async_b128(offV + (unsigned)(r2 * 64 + c0) * 2, gV + (size_t)r2 * SLEN + c0);
      }
      const int p0 = SLEN - i0 - 63 + j0;  // in (0, 2S-128], no clamp needed
      const bf16* gR = Krp + (size_t)p0 * HDIM;
      for (int idx = tid; idx < 1024; idx += 128) {
        int j = idx >> 3, d0 = (idx & 7) * 8;
        async_b128(offKr + (unsigned)frag_off(j, d0) * 2, gR + idx * 8);
      }
    }
    wait_async0();
    __syncthreads();

    // ---- AC = Q_ac . K^T (64x64): B fragment = one 32B LDS load
#pragma unroll
    for (int t = 0; t < 4; ++t) {
      v8f acc;
#pragma unroll
      for (int z = 0; z < 8; ++z) acc[z] = 0.f;
#pragma unroll
      for (int ks = 0; ks < 2; ++ks)
        acc = wmma_bf16(a_ac[ks], frag32(&sKP[((t * 2 + ks) * 32 + lane) * 16]),
                        acc);
#pragma unroll
      for (int v = 0; v < 8; ++v)
        sS[(16 * wave + v + Lhi8) * 64 + 16 * t + (lane & 15)] = (bf16)acc[v];
    }
    // ---- BD = Q_bd . Kr_window^T (64x128)
#pragma unroll
    for (int t = 0; t < 8; ++t) {
      v8f acc;
#pragma unroll
      for (int z = 0; z < 8; ++z) acc[z] = 0.f;
#pragma unroll
      for (int ks = 0; ks < 2; ++ks)
        acc = wmma_bf16(a_bd[ks], frag32(&sKr[((t * 2 + ks) * 32 + lane) * 16]),
                        acc);
#pragma unroll
      for (int v = 0; v < 8; ++v)
        sBD[(16 * wave + v + Lhi8) * 128 + 16 * t + (lane & 15)] = (bf16)acc[v];
    }
    __syncthreads();

    // ---- score combine + online softmax (one thread per row)
    if (tid < 64) {
      const int li = tid, gi = i0 + li;
      const int shift = 63 - li;  // rel_shift: bd[i,j] = BD[li][(63-li)+jl]
      const float ef0 = ef[((size_t)bn * SLEN + gi) * 2 + 0];
      const float ef1 = ef[((size_t)bn * SLEN + gi) * 2 + 1];
      float mold = sM[li], mnew = mold;
      float sc[64];
#pragma unroll
      for (int jl = 0; jl < 64; ++jl) {
        int gj = j0 + jl;
        float s = (float)sS[li * 64 + jl] + (float)sBD[li * 128 + shift + jl];
        s += seg_mat[((size_t)gi * SLEN + gj) * BDIM + b] ? ef1 : ef0;
        s = s * SCALE_F - 1.0e30f * mask[((size_t)gi * SLEN + gj) * BDIM + b];
        sc[jl] = s;
        mnew = fmaxf(mnew, s);
      }
      float fac = __expf(mold - mnew);
      float l = sL[li] * fac;
#pragma unroll
      for (int jl = 0; jl < 64; ++jl) {
        float p = __expf(sc[jl] - mnew);
        l += p;
        sKP[li * 64 + jl] = (bf16)p;  // P tile row-major (reuses K region)
      }
      sM[li] = mnew; sL[li] = l; sFac[li] = fac;
    }
    __syncthreads();

    // ---- rescale O and accumulate P.V
    float frow[8];
#pragma unroll
    for (int v = 0; v < 8; ++v) frow[v] = sFac[16 * wave + v + Lhi8];
    v16bf pa[2];
#pragma unroll
    for (int ks = 0; ks < 2; ++ks) {
      const bf16* p = &sKP[(16 * wave + (lane & 15)) * 64 + Lhi16 + 32 * ks];
      pa[ks] = frag_from2(p, p + 8);
    }
#pragma unroll
    for (int t = 0; t < 4; ++t) {
#pragma unroll
      for (int v = 0; v < 8; ++v) o[t][v] *= frow[v];
#pragma unroll
      for (int ks = 0; ks < 2; ++ks)
        o[t] = wmma_bf16(
            pa[ks],
            frag32(&sVt[(16 * t + (lane & 15)) * 64 + Lhi16 + 32 * ks]), o[t]);
    }
    __syncthreads();
  }

  // ---- normalize and store attn_vec (bf16, [b][n][i][h])
#pragma unroll
  for (int t = 0; t < 4; ++t) {
#pragma unroll
    for (int v = 0; v < 8; ++v) {
      int li = 16 * wave + v + Lhi8;
      float inv = 1.f / sL[li];
      attn_out[headoff + (size_t)(i0 + li) * HDIM + 16 * t + (lane & 15)] =
          (bf16)(o[t][v] * inv);
    }
  }
}

// ============================================================================
// Output projection: out[i*B+b][hD] = sum_{k=n*H+d} attn[b][n][i][d]*projC[hD][k]
// projC is proj_o converted to bf16 in native [hD][k] layout (col-fragment-ready)
// ============================================================================
__global__ __launch_bounds__(128)
void outproj_gemm_kernel(const bf16* __restrict__ A,      // [B][N][S][H]
                         const bf16* __restrict__ projC,  // [D][N*H]
                         float* __restrict__ out) {       // [S*B][D]
  __shared__ __align__(32) bf16 ldsA[64 * 32];  // [m][k]
  __shared__ __align__(32) bf16 ldsW[64 * 32];  // [hD_local][k]
  const int tid = threadIdx.x, wave = tid >> 5, lane = tid & 31;
  const int m0 = blockIdx.x * 64, n0 = blockIdx.y * 64;
  const int Lhi8  = (lane >= 16) ? 8 : 0;
  const int Lhi16 = (lane >= 16) ? 16 : 0;
  const int ml = 16 * wave + (lane & 15);

  v8f acc[4];
#pragma unroll
  for (int t = 0; t < 4; ++t)
#pragma unroll
    for (int z = 0; z < 8; ++z) acc[t][z] = 0.f;

  for (int k0 = 0; k0 < NHEAD * HDIM; k0 += 32) {
    const int n = k0 >> 6, dbase = k0 & 63;  // 32-wide K slab stays in one head
    for (int idx = tid; idx < 256; idx += 128) {
      int r2 = idx >> 2, c8 = (idx & 3) * 8;
      int m = m0 + r2;
      int i = m / BDIM, bb = m % BDIM;
      *(uint4*)&ldsA[r2 * 32 + c8] =
          *(const uint4*)&A[(((size_t)(bb * NHEAD + n)) * SLEN + i) * HDIM +
                            dbase + c8];
    }
    for (int idx = tid; idx < 256; idx += 128) {
      int nl = idx >> 2, c8 = (idx & 3) * 8;
      *(uint4*)&ldsW[nl * 32 + c8] =
          *(const uint4*)&projC[(size_t)(n0 + nl) * (NHEAD * HDIM) + k0 + c8];
    }
    __syncthreads();
    v16bf a;
    {
      const bf16* p = &ldsA[ml * 32 + Lhi16];
      a = frag_from2(p, p + 8);
    }
#pragma unroll
    for (int t = 0; t < 4; ++t)
      acc[t] = wmma_bf16(a, frag32(&ldsW[(16 * t + (lane & 15)) * 32 + Lhi16]),
                         acc[t]);
    __syncthreads();
  }
#pragma unroll
  for (int t = 0; t < 4; ++t)
#pragma unroll
    for (int v = 0; v < 8; ++v)
      out[(size_t)(m0 + 16 * wave + v + Lhi8) * DDIM + n0 + 16 * t +
          (lane & 15)] = acc[t][v];
}

// ============================================================================
// Residual + LayerNorm over D=1024, one block (256 threads) per (i,b) row.
// ============================================================================
__global__ __launch_bounds__(256)
void ln_kernel(const float* __restrict__ x, const float* __restrict__ res,
               const float* __restrict__ gamma, const float* __restrict__ beta,
               float* __restrict__ out) {
  __shared__ float rsum[256], rsq[256];
  const int row = blockIdx.x, tid = threadIdx.x;
  float s = 0.f, s2 = 0.f;
  for (int c = tid; c < DDIM; c += 256) {
    float v = x[(size_t)row * DDIM + c] + res[(size_t)row * DDIM + c];
    s += v; s2 += v * v;
  }
  rsum[tid] = s; rsq[tid] = s2;
  __syncthreads();
  for (int st = 128; st > 0; st >>= 1) {
    if (tid < st) { rsum[tid] += rsum[tid + st]; rsq[tid] += rsq[tid + st]; }
    __syncthreads();
  }
  float mu  = rsum[0] * (1.f / DDIM);
  float var = rsq[0] * (1.f / DDIM) - mu * mu;
  float inv = rsqrtf(var + 1e-12f);
  for (int c = tid; c < DDIM; c += 256) {
    float v = x[(size_t)row * DDIM + c] + res[(size_t)row * DDIM + c];
    out[(size_t)row * DDIM + c] = (v - mu) * inv * gamma[c] + beta[c];
  }
}

// ============================================================================
extern "C" void kernel_launch(void* const* d_in, const int* in_sizes, int n_in,
                              void* d_out, int out_size, void* d_ws, size_t ws_size,
                              hipStream_t stream) {
  (void)in_sizes; (void)n_in; (void)out_size; (void)ws_size;
  const float* h        = (const float*)d_in[0];
  const float* g        = (const float*)d_in[1];
  const float* r        = (const float*)d_in[2];
  const unsigned char* seg = (const unsigned char*)d_in[3];
  const float* mask_h   = (const float*)d_in[4];
  const float* mask_g   = (const float*)d_in[5];
  const float* q_w      = (const float*)d_in[6];
  const float* k_w      = (const float*)d_in[7];
  const float* v_w      = (const float*)d_in[8];
  const float* r_w      = (const float*)d_in[9];
  const float* proj_o   = (const float*)d_in[10];
  const float* r_w_bias = (const float*)d_in[11];
  const float* r_r_bias = (const float*)d_in[12];
  const float* r_s_bias = (const float*)d_in[13];
  const float* seg_emb  = (const float*)d_in[14];
  const float* ln_gamma = (const float*)d_in[15];
  const float* ln_beta  = (const float*)d_in[16];
  float* out = (float*)d_out;

  // ---- workspace carve-up (256B aligned; total ~77 MB)
  char* ws = (char*)d_ws;
  size_t off = 0;
  auto alloc = [&](size_t bytes) -> char* {
    char* p = ws + off;
    off += (bytes + 255) & ~(size_t)255;
    return p;
  };
  const size_t headElems = (size_t)BDIM * NHEAD * SLEN * HDIM;  // 2M
  const size_t wElems    = (size_t)DDIM * NHEAD * HDIM;         // 1M
  bf16* wqT   = (bf16*)alloc(wElems * 2);
  bf16* wkT   = (bf16*)alloc(wElems * 2);
  bf16* wvT   = (bf16*)alloc(wElems * 2);
  bf16* wrT   = (bf16*)alloc(wElems * 2);
  bf16* projC = (bf16*)alloc(wElems * 2);
  bf16* kh    = (bf16*)alloc(headElems * 2);
  bf16* vT    = (bf16*)alloc(headElems * 2);  // per-head transposed V [b][n][h][i]
  bf16* kr    = (bf16*)alloc((size_t)BDIM * NHEAD * RLEN * HDIM * 2);
  bf16* qac_h = (bf16*)alloc(headElems * 2);
  bf16* qbd_h = (bf16*)alloc(headElems * 2);
  bf16* qsb_h = (bf16*)alloc(headElems * 2);
  bf16* qac_g = (bf16*)alloc(headElems * 2);
  bf16* qbd_g = (bf16*)alloc(headElems * 2);
  bf16* qsb_g = (bf16*)alloc(headElems * 2);
  float* ef_h = (float*)alloc((size_t)BDIM * NHEAD * SLEN * 2 * 4);
  float* ef_g = (float*)alloc((size_t)BDIM * NHEAD * SLEN * 2 * 4);
  bf16* av_h  = (bf16*)alloc(headElems * 2);
  bf16* av_g  = (bf16*)alloc(headElems * 2);
  float* po_h = (float*)alloc((size_t)SLEN * BDIM * DDIM * 4);
  float* po_g = (float*)alloc((size_t)SLEN * BDIM * DDIM * 4);

  const dim3 blk128(128);
  const dim3 gW((DDIM * NHEAD * HDIM) / 256);

  // weight prep (transpose+convert to bf16)
  transpose_w_kernel<<<gW, 256, 0, stream>>>(q_w, wqT);
  transpose_w_kernel<<<gW, 256, 0, stream>>>(k_w, wkT);
  transpose_w_kernel<<<gW, 256, 0, stream>>>(v_w, wvT);
  transpose_w_kernel<<<gW, 256, 0, stream>>>(r_w, wrT);
  convert_w_kernel<<<gW, 256, 0, stream>>>(proj_o, projC);

  // projections (bf16 WMMA)
  const dim3 gProj((SLEN * BDIM) / 64, (NHEAD * HDIM) / 64);
  const dim3 gProjR((RLEN * BDIM) / 64, (NHEAD * HDIM) / 64);
  proj_gemm_kernel<<<gProj, blk128, 0, stream>>>(
      h, wkT, SLEN, kh, nullptr, nullptr, nullptr, nullptr, nullptr, nullptr);
  proj_gemm_kernel<<<gProj, blk128, 0, stream>>>(
      h, wvT, SLEN, nullptr, nullptr, nullptr, nullptr, nullptr, nullptr, vT);
  proj_gemm_kernel<<<gProj, blk128, 0, stream>>>(
      h, wqT, SLEN, qac_h, r_w_bias, qbd_h, r_r_bias, qsb_h, r_s_bias, nullptr);
  proj_gemm_kernel<<<gProj, blk128, 0, stream>>>(
      g, wqT, SLEN, qac_g, r_w_bias, qbd_g, r_r_bias, qsb_g, r_s_bias, nullptr);
  proj_gemm_kernel<<<gProjR, blk128, 0, stream>>>(
      r, wrT, RLEN, kr, nullptr, nullptr, nullptr, nullptr, nullptr, nullptr);

  ef_kernel<<<dim3((BDIM * NHEAD * SLEN) / 256), 256, 0, stream>>>(qsb_h, seg_emb, ef_h);
  ef_kernel<<<dim3((BDIM * NHEAD * SLEN) / 256), 256, 0, stream>>>(qsb_g, seg_emb, ef_g);

  // fused flash relative attention (h and g streams)
  const dim3 gAttn(SLEN / 64, BDIM * NHEAD);
  attn_kernel<<<gAttn, blk128, 0, stream>>>(qac_h, qbd_h, kh, vT, kr, ef_h,
                                            seg, mask_h, av_h);
  attn_kernel<<<gAttn, blk128, 0, stream>>>(qac_g, qbd_g, kh, vT, kr, ef_g,
                                            seg, mask_g, av_g);

  // output projection + residual LayerNorm
  const dim3 gOut((SLEN * BDIM) / 64, DDIM / 64);
  outproj_gemm_kernel<<<gOut, blk128, 0, stream>>>(av_h, projC, po_h);
  outproj_gemm_kernel<<<gOut, blk128, 0, stream>>>(av_g, projC, po_g);
  ln_kernel<<<dim3(SLEN * BDIM), 256, 0, stream>>>(po_h, h, ln_gamma, ln_beta, out);
  ln_kernel<<<dim3(SLEN * BDIM), 256, 0, stream>>>(po_g, g, ln_gamma, ln_beta,
                                                   out + (size_t)SLEN * BDIM * DDIM);
}
